// ParallelAttention_66236985639280
// MI455X (gfx1250) — compile-verified, matrix operand order checked by reading, not run on hardware
//
#include <hip/hip_runtime.h>
#include <hip/hip_bf16.h>

// ---------------------------------------------------------------------------
// ParallelAttention forward for MI455X (gfx1250), wave32 + WMMA + async LDS.
//   B=4, L=2048, D=256, H=8, HD=32
// GEMMs: v_wmma_f32_16x16x32_f16 (f16 in, fp32 accumulate).
// Attention: flash-style streaming softmax; K/V tiles staged into LDS with
// global_load_async_to_lds_b128 (ASYNCcnt double buffering), consumed via
// ds_load_b128 fragment loads shared by all 8 waves of the workgroup.
// ---------------------------------------------------------------------------

#define Bc  4
#define Lc  2048
#define Dc  256
#define Hc  8
#define HDc 32
#define BLD (Bc * Lc * Dc)          // 2,097,152
#define SCALE 0.17677669529663687f  // 1/sqrt(32)

typedef __attribute__((ext_vector_type(16))) _Float16 v16h;
typedef __attribute__((ext_vector_type(8)))  _Float16 v8h;
typedef __attribute__((ext_vector_type(8)))  float    v8f;

// A/B fragment loader for v_wmma_f32_16x16x32_f16.
// ISA 16-bit A layout (16x32, MxK): lane L -> row M = L%16; half = L/16.
//   regs 0..3 hold K = half*8 + 0..7, regs 4..7 hold K = 16 + half*8 + 0..7.
// Caller passes p = &Mat[row][half*8]; reads 8 halves at p and 8 at p+16.
// B operand assumed lane->N-column symmetric, so B is loaded with the same
// routine from a column-major (transposed) matrix.
__device__ __forceinline__ v16h load_ab_frag(const _Float16* p) {
  v8h lo = *reinterpret_cast<const v8h*>(p);
  v8h hi = *reinterpret_cast<const v8h*>(p + 16);
  v16h f;
#pragma unroll
  for (int i = 0; i < 8; ++i) { f[i] = lo[i]; f[i + 8] = hi[i]; }
  return f;
}

__device__ __forceinline__ v8f wmma_f16(v16h a, v16h b, v8f c) {
  return __builtin_amdgcn_wmma_f32_16x16x32_f16(false, a, false, b,
                                                (short)0, c, false, false);
}

// CDNA5 async global->LDS copy: per lane, 16 bytes mem -> LDS (ASYNCcnt).
__device__ __forceinline__ void async_copy_b128(unsigned int lds_addr,
                                                unsigned long long gaddr) {
  asm volatile("global_load_async_to_lds_b128 %0, %1, off"
               :: "v"(lds_addr), "v"(gaddr) : "memory");
}
#define WAIT_ASYNCCNT(N) \
  asm volatile("s_wait_asynccnt %0" :: "n"(N) : "memory")

// ---------------------------------------------------------------------------
// Stage 0: fp32 -> f16 conversions
// ---------------------------------------------------------------------------
__global__ void __launch_bounds__(256) cvt_f32_f16(const float* __restrict__ src,
                                                   _Float16* __restrict__ dst, int n) {
  int i = blockIdx.x * 256 + threadIdx.x;
  if (i < n) dst[i] = (_Float16)src[i];
}

// W[K][N] (fp32) -> WT[N][K] (f16): makes B-fragment loads contiguous.
__global__ void __launch_bounds__(256) cvt_transpose(const float* __restrict__ W,
                                                     _Float16* __restrict__ WT,
                                                     int K, int N) {
  int i = blockIdx.x * 256 + threadIdx.x;
  if (i < K * N) {
    int k = i / N, n = i % N;
    WT[(size_t)n * K + k] = (_Float16)W[i];
  }
}

// ---------------------------------------------------------------------------
// Stage 1: qkv = x @ Wqkv, scattered into per-head Q,K (row-major [l][hd])
// and V transposed ([hd][l]).  One wave per 16x16 output tile.
// ---------------------------------------------------------------------------
__global__ void __launch_bounds__(256) qkv_gemm(const _Float16* __restrict__ xh,
                                                const _Float16* __restrict__ WT,
                                                _Float16* __restrict__ qbuf,
                                                _Float16* __restrict__ kbuf,
                                                _Float16* __restrict__ vT) {
  int wid  = blockIdx.x * 8 + (threadIdx.x >> 5);
  int lane = threadIdx.x & 31;
  int r = lane & 15, half = lane >> 4;
  int tm = wid / 48, tn = wid % 48;

  const _Float16* ap = xh + (size_t)(tm * 16 + r) * Dc + half * 8;
  const _Float16* bp = WT + (size_t)(tn * 16 + r) * Dc + half * 8;

  v8f acc = {};
#pragma unroll
  for (int kk = 0; kk < Dc; kk += 32)
    acc = wmma_f16(load_ab_frag(ap + kk), load_ab_frag(bp + kk), acc);

  int n   = tn * 16 + r;          // 0..767
  int sec = n >> 8;               // 0=Q 1=K 2=V
  int nn  = n & 255;
  int h   = nn >> 5, hd = nn & 31;
  int m0  = tm * 16 + 8 * half;
#pragma unroll
  for (int e = 0; e < 8; ++e) {
    int m = m0 + e;
    int b = m >> 11;
    int l = m & 2047;
    _Float16 hv = (_Float16)acc[e];
    size_t bh = (size_t)b * Hc + h;
    if (sec == 0)      qbuf[(bh * Lc + l) * HDc + hd] = hv;
    else if (sec == 1) kbuf[(bh * Lc + l) * HDc + hd] = hv;
    else               vT[(bh * HDc + hd) * Lc + l]   = hv;
  }
}

// ---------------------------------------------------------------------------
// Stage 2: flash attention. Block = 8 waves sharing one (b,h); wave w owns
// i-tile (group*8+w). Per 32-wide j-step: K/V tiles arrive in LDS via async
// copies (double buffered), S = Q K^T (2 WMMA), online softmax in per-wave
// LDS scratch, O += P V (2 WMMA).
// ---------------------------------------------------------------------------
__global__ void __launch_bounds__(256) attn_kernel(const _Float16* __restrict__ qbuf,
                                                   const _Float16* __restrict__ kbuf,
                                                   const _Float16* __restrict__ vT,
                                                   const float* __restrict__ bias,
                                                   _Float16* __restrict__ aout) {
  __shared__ _Float16 Ktile[2][32][32];   // [slot][j][hd]    2KB/slot
  __shared__ _Float16 Vtile[2][32][32];   // [slot][hd][j]    2KB/slot
  __shared__ float    Ssm[8][16][33];     // per-wave score tile
  __shared__ _Float16 Psm[8][16][32];     // per-wave P = exp(S-m), f16
  __shared__ float    mrow[8][16], lrow[8][16], arow[8][16];

  int w    = threadIdx.x >> 5;
  int lane = threadIdx.x & 31;
  int r = lane & 15, half = lane >> 4;
  int bh   = blockIdx.x >> 4;             // 0..31 = b*8+h
  int grp  = blockIdx.x & 15;             // 16 groups of 8 i-tiles
  int b    = bh >> 3, h = bh & 7;
  int i0   = (grp * 8 + w) * 16;

  const _Float16* kb = kbuf + (size_t)bh * Lc * HDc;  // [l][hd], rows contig.
  const _Float16* vb = vT   + (size_t)bh * HDc * Lc;  // [hd][l]
  unsigned int ldsK = (unsigned int)(size_t)&Ktile[0][0][0];
  unsigned int ldsV = (unsigned int)(size_t)&Vtile[0][0][0];

  // Async tile copy issuers (wave0: K, wave1: V). 4 x 512B per tile.
  auto issueK = [&](int slot, int jt) {
    unsigned long long g = (unsigned long long)(size_t)(kb + (size_t)jt * HDc);
#pragma unroll
    for (int t = 0; t < 4; ++t) {
      unsigned int idx = (unsigned int)(t * 512 + lane * 16);
      async_copy_b128(ldsK + (unsigned int)slot * 2048u + idx, g + idx);
    }
  };
  auto issueV = [&](int slot, int jt) {
#pragma unroll
    for (int t = 0; t < 4; ++t) {
      int hd = t * 8 + (lane >> 2);
      int ch = (lane & 3) * 16;           // byte chunk within 64B row
      unsigned long long g =
          (unsigned long long)(size_t)(vb + (size_t)hd * Lc + jt) + (unsigned)ch;
      unsigned int l = ldsV + (unsigned int)slot * 2048u +
                       (unsigned int)(t * 512 + lane * 16);
      async_copy_b128(l, g);
    }
  };

  // Q fragment, reused for all 64 j-steps.
  v16h aq = load_ab_frag(qbuf + ((size_t)bh * Lc + i0 + r) * HDc + half * 8);

  v8f acc0 = {};  // out columns hd 0..15
  v8f acc1 = {};  // out columns hd 16..31
  if (lane < 16) { mrow[w][lane] = -1e30f; lrow[w][lane] = 0.f; }

  // Prologue: fill both buffers.
  if (w == 0) { issueK(0, 0); issueK(1, 32); }
  if (w == 1) { issueV(0, 0); issueV(1, 32); }

  for (int jt = 0; jt < Lc; jt += 32) {
    int cur = (jt >> 5) & 1;
    WAIT_ASYNCCNT(4);            // issuers: oldest tile (slot cur) landed
    __syncthreads();             // B1: tile visible to every wave

    // Fragments from LDS (ds_load_b128).
    v16h bk0 = load_ab_frag(&Ktile[cur][r][half * 8]);
    v16h bk1 = load_ab_frag(&Ktile[cur][16 + r][half * 8]);
    v16h bv0 = load_ab_frag(&Vtile[cur][r][half * 8]);
    v16h bv1 = load_ab_frag(&Vtile[cur][16 + r][half * 8]);

    v8f s0 = {}, s1 = {};
    s0 = wmma_f16(aq, bk0, s0);
    s1 = wmma_f16(aq, bk1, s1);
#pragma unroll
    for (int e = 0; e < 8; ++e) {
      Ssm[w][e + 8 * half][r]      = s0[e];
      Ssm[w][e + 8 * half][16 + r] = s1[e];
    }
    __syncthreads();             // B2: Ssm ready; tile reads retired

    // Overlap: refill slot `cur` for tile jt+64 while softmax runs.
    if (jt + 64 < Lc) {
      if (w == 0) issueK(cur, jt + 64);
      if (w == 1) issueV(cur, jt + 64);
    }

    // Online softmax: lanes 0..15 each own one row of the tile.
    if (lane < 16) {
      const float* bp = bias + (size_t)(i0 + lane) * Lc + jt;
      float mloc = -1e30f;
#pragma unroll
      for (int c = 0; c < 32; ++c) {
        float v = Ssm[w][lane][c] * SCALE + bp[c];
        Ssm[w][lane][c] = v;
        mloc = fmaxf(mloc, v);
      }
      float mold = mrow[w][lane];
      float mnew = fmaxf(mold, mloc);
      float al   = __expf(mold - mnew);
      float sum  = 0.f;
#pragma unroll
      for (int c = 0; c < 32; ++c) {
        float pe = __expf(Ssm[w][lane][c] - mnew);
        Psm[w][lane][c] = (_Float16)pe;
        sum += pe;
      }
      mrow[w][lane] = mnew;
      lrow[w][lane] = lrow[w][lane] * al + sum;
      arow[w][lane] = al;
    }
    __syncthreads();             // B3: Psm/stats ready

    // Rescale running output, then O += P V.
#pragma unroll
    for (int e = 0; e < 8; ++e) {
      float al = arow[w][e + 8 * half];
      acc0[e] *= al;
      acc1[e] *= al;
    }
    v16h ap = load_ab_frag(&Psm[w][r][half * 8]);
    acc0 = wmma_f16(ap, bv0, acc0);
    acc1 = wmma_f16(ap, bv1, acc1);
  }

  // Normalize and store merged heads: aout[b][l][h*32+hd] (f16).
#pragma unroll
  for (int e = 0; e < 8; ++e) {
    int rl = e + 8 * half;
    int l  = i0 + rl;
    float inv = 1.0f / lrow[w][rl];
    size_t o = ((size_t)b * Lc + l) * Dc + h * HDc;
    aout[o + r]      = (_Float16)(acc0[e] * inv);
    aout[o + 16 + r] = (_Float16)(acc1[e] * inv);
  }
}

// ---------------------------------------------------------------------------
// Stage 3: out = attn_out @ Wout  (M=8192, N=256, K=256) -> fp32
// ---------------------------------------------------------------------------
__global__ void __launch_bounds__(256) out_gemm(const _Float16* __restrict__ A,
                                                const _Float16* __restrict__ BT,
                                                float* __restrict__ out) {
  int wid  = blockIdx.x * 8 + (threadIdx.x >> 5);
  int lane = threadIdx.x & 31;
  int r = lane & 15, half = lane >> 4;
  int tm = wid >> 4, tn = wid & 15;

  const _Float16* ap = A  + (size_t)(tm * 16 + r) * Dc + half * 8;
  const _Float16* bp = BT + (size_t)(tn * 16 + r) * Dc + half * 8;

  v8f acc = {};
#pragma unroll
  for (int kk = 0; kk < Dc; kk += 32)
    acc = wmma_f16(load_ab_frag(ap + kk), load_ab_frag(bp + kk), acc);

  int n  = tn * 16 + r;
  int m0 = tm * 16 + 8 * half;
#pragma unroll
  for (int e = 0; e < 8; ++e)
    out[(size_t)(m0 + e) * Dc + n] = acc[e];
}

// ---------------------------------------------------------------------------
extern "C" void kernel_launch(void* const* d_in, const int* in_sizes, int n_in,
                              void* d_out, int out_size, void* d_ws, size_t ws_size,
                              hipStream_t stream) {
  const float* x    = (const float*)d_in[0];
  const float* bias = (const float*)d_in[1];
  const float* Wqkv = (const float*)d_in[2];
  const float* Wout = (const float*)d_in[3];
  float* out = (float*)d_out;

  char* p = (char*)d_ws;
  auto carve = [&](size_t bytes) {
    void* q = (void*)p;
    p += (bytes + 255) & ~(size_t)255;
    return q;
  };
  _Float16* xh    = (_Float16*)carve((size_t)BLD * 2);        // x as f16
  _Float16* WqkvT = (_Float16*)carve((size_t)768 * 256 * 2);  // [3D][D]
  _Float16* WoutT = (_Float16*)carve((size_t)256 * 256 * 2);  // [D][D]
  _Float16* qbuf  = (_Float16*)carve((size_t)BLD * 2);        // [b][h][l][hd]
  _Float16* kbuf  = (_Float16*)carve((size_t)BLD * 2);        // [b][h][l][hd]
  _Float16* vTb   = (_Float16*)carve((size_t)BLD * 2);        // [b][h][hd][l]
  _Float16* aout  = (_Float16*)carve((size_t)BLD * 2);        // [b][l][d]

  cvt_f32_f16 <<<BLD / 256, 256, 0, stream>>>(x, xh, BLD);
  cvt_transpose<<<(256 * 768) / 256, 256, 0, stream>>>(Wqkv, WqkvT, 256, 768);
  cvt_transpose<<<(256 * 256) / 256, 256, 0, stream>>>(Wout, WoutT, 256, 256);

  qkv_gemm   <<<24576 / 8, 256, 0, stream>>>(xh, WqkvT, qbuf, kbuf, vTb);
  attn_kernel<<<512,       256, 0, stream>>>(qbuf, kbuf, vTb, bias, aout);
  out_gemm   <<<8192 / 8,  256, 0, stream>>>(aout, WoutT, out);

  (void)in_sizes; (void)n_in; (void)out_size; (void)ws_size;
}
